// ManifoldLayer_4045859193505
// MI455X (gfx1250) — matrix operand order repaired and checked by reading, not compiled, not run on hardware
//
#include <hip/hip_runtime.h>
#include <hip/hip_bf16.h>

typedef __attribute__((ext_vector_type(16))) __bf16 v16bf;
typedef __attribute__((ext_vector_type(8)))  __bf16 v8bf;
typedef __attribute__((ext_vector_type(8)))  float  v8f;

#define N_PTS 1024
#define DIM   128
#define OUTD  256
#define XS    136   // padded LDS row stride (bf16 elems) -> 272B, bank-conflict free

// ---------------- kernel 1: metric = M @ M^T (fp32, tiny) ----------------
__global__ void k_metric(const float* __restrict__ M, float* __restrict__ metric) {
    int idx = blockIdx.x * blockDim.x + threadIdx.x;   // 0..16383
    int r = idx >> 7, c = idx & 127;
    const float* mr = M + r * DIM;
    const float* mc = M + c * DIM;
    float acc = 0.f;
    #pragma unroll 8
    for (int k = 0; k < DIM; ++k) acc += mr[k] * mc[k];
    metric[idx] = acc;
}

// ---------------- kernel 2: xm = x @ metric (fp32 + bf16 copy) ----------------
__global__ void k_xm(const float* __restrict__ x, const float* __restrict__ metric,
                     float* __restrict__ xmf, __bf16* __restrict__ xmb) {
    int idx = blockIdx.x * blockDim.x + threadIdx.x;   // 0..131071
    int n = idx >> 7, d = idx & 127;
    const float* xr = x + n * DIM;
    float acc = 0.f;
    #pragma unroll 8
    for (int k = 0; k < DIM; ++k) acc += xr[k] * metric[k * DIM + d];
    xmf[idx] = acc;
    xmb[idx] = (__bf16)acc;
}

// ------- kernel 3: swizzle connection [16384 x 128] into bf16 WMMA B-panels -------
// Fragment f = chunk*8 + ktile holds C_flat[chunk*32 .. +31, ktile*16 .. +15].
// Within a fragment: element (lane,r) = C_flat[chunk*32 + (lane/16)*16 + r,
//                                             ktile*16 + lane%16]
// so each lane's 16 bf16 (32B) are contiguous -> coalesced 2x b128 loads.
__global__ void k_swz_conn(const float* __restrict__ conn, __bf16* __restrict__ sw) {
    int e = blockIdx.x * blockDim.x + threadIdx.x;     // 0..2097151
    int f = e >> 9, s = e & 511;
    int lane = s >> 4, r = s & 15;
    int c = f >> 3, t = f & 7;
    int K = c * 32 + ((lane >> 4) << 4) + r;           // flattened (i,j)
    int n = t * 16 + (lane & 15);                      // k
    sw[e] = (__bf16)conn[K * DIM + n];
}

// ------- kernel 4: swizzle proj_w^T [128 x 256] into bf16 WMMA B-panels -------
__global__ void k_swz_pw(const float* __restrict__ w, __bf16* __restrict__ sw) {
    int e = blockIdx.x * blockDim.x + threadIdx.x;     // 0..32767
    int f = e >> 9, s = e & 511;
    int lane = s >> 4, r = s & 15;
    int c = f >> 4, t = f & 15;
    int k = c * 32 + ((lane >> 4) << 4) + r;
    int o = t * 16 + (lane & 15);
    sw[e] = (__bf16)w[o * DIM + k];                    // W[o,k], B = W^T
}

// ---------------- kernel 5: fused bilinear (WMMA) + projection (WMMA) ----------------
// 64 workgroups x 256 threads (8 waves). Workgroup = 16 batches, full k.
// Wave w: acc[16x16] over K=16384 in 512 x wmma_f32_16x16x32_bf16, A built on the fly.
__launch_bounds__(256, 1)
__global__ void k_main(const float* __restrict__ xmf, const __bf16* __restrict__ xmb,
                       const __bf16* __restrict__ connsw, const __bf16* __restrict__ pwsw,
                       const float* __restrict__ bias, float* __restrict__ out) {
    __shared__ __align__(16) __bf16 sXm[16 * XS];
    __shared__ __align__(16) __bf16 sT[16 * XS];
    const int tid  = threadIdx.x;
    const int wave = tid >> 5;          // wave32
    const int lane = tid & 31;
    const int row  = lane & 15;
    const int hi   = lane >> 4;
    const int btile = blockIdx.x;

    // stage xm tile (16 x 128) into LDS as bf16
    for (int e = tid; e < 16 * DIM; e += 256) {
        int rr = e >> 7, cc = e & 127;
        sXm[rr * XS + cc] = xmb[(btile * 16 + rr) * DIM + cc];
    }
    __syncthreads();

    const __bf16* aRow = sXm + row * XS + hi * 16;
    const int kt = wave;                               // k-tile 0..7
    v8f acc = {};

    for (int i = 0; i < DIM; ++i) {
        float xi = (float)sXm[row * XS + i];
        #pragma unroll
        for (int jc = 0; jc < 4; ++jc) {
            // A fragment: xm[b,i] * xm[b, j0..j0+31] in 16-bit A VGPR layout
            v8bf jl = *(const v8bf*)(aRow + jc * 32);
            v8bf jh = *(const v8bf*)(aRow + jc * 32 + 8);
            v16bf a;
            #pragma unroll
            for (int r = 0; r < 8; ++r) {
                a[r]     = (__bf16)((float)jl[r] * xi);
                a[r + 8] = (__bf16)((float)jh[r] * xi);
            }
            int chunk = i * 4 + jc;
            const __bf16* bp = connsw + ((size_t)(chunk * 8 + kt) << 9) + lane * 16;
            v16bf b = *(const v16bf*)bp;               // 2x global_load_b128, coalesced
            acc = __builtin_amdgcn_wmma_f32_16x16x32_bf16(false, a, false, b,
                                                          (short)0, acc, false, false);
        }
    }

    // transported = xm + bilinear; stash tile as bf16 for projection GEMM
    #pragma unroll
    for (int p = 0; p < 8; ++p) {
        int m = hi * 8 + p;                            // C/D layout: M = p + hi*8
        int col = kt * 16 + row;                       // N = lane%16
        float t = acc[p] + xmf[(btile * 16 + m) * DIM + col];
        sT[m * XS + col] = (__bf16)t;
    }
    __syncthreads();

    // projection: out[16 x 256] tile, 16 o-tiles across 8 waves (2 each), K=128
    const __bf16* tRow = sT + row * XS + hi * 16;
    #pragma unroll
    for (int rep = 0; rep < 2; ++rep) {
        int ot = wave * 2 + rep;
        v8f pacc = {};
        #pragma unroll
        for (int c2 = 0; c2 < 4; ++c2) {
            v8bf jl = *(const v8bf*)(tRow + c2 * 32);
            v8bf jh = *(const v8bf*)(tRow + c2 * 32 + 8);
            v16bf a;
            #pragma unroll
            for (int r = 0; r < 8; ++r) { a[r] = jl[r]; a[r + 8] = jh[r]; }
            const __bf16* bp = pwsw + ((size_t)(c2 * 16 + ot) << 9) + lane * 16;
            v16bf b = *(const v16bf*)bp;
            pacc = __builtin_amdgcn_wmma_f32_16x16x32_bf16(false, a, false, b,
                                                           (short)0, pacc, false, false);
        }
        #pragma unroll
        for (int p = 0; p < 8; ++p) {
            int m = hi * 8 + p;
            int o = ot * 16 + row;
            out[(btile * 16 + m) * OUTD + o] = pacc[p] + bias[o];
        }
    }
}

extern "C" void kernel_launch(void* const* d_in, const int* in_sizes, int n_in,
                              void* d_out, int out_size, void* d_ws, size_t ws_size,
                              hipStream_t stream) {
    (void)in_sizes; (void)n_in; (void)out_size; (void)ws_size;
    const float* x    = (const float*)d_in[0];   // [1024,128]
    const float* mt   = (const float*)d_in[1];   // [128,128]
    const float* conn = (const float*)d_in[2];   // [128,128,128]
    const float* pw   = (const float*)d_in[3];   // [256,128]
    const float* pb   = (const float*)d_in[4];   // [256]
    float* out = (float*)d_out;                  // [1024,256] fp32

    // workspace layout (all offsets 32B-aligned), total ~4.9 MB
    char* ws = (char*)d_ws;
    float*  metric = (float*)(ws);                                   // 64 KB
    float*  xmf    = (float*)(ws + 65536);                           // 512 KB
    __bf16* xmb    = (__bf16*)(ws + 65536 + 524288);                 // 256 KB
    __bf16* connsw = (__bf16*)(ws + 65536 + 524288 + 262144);        // 4 MB
    __bf16* pwsw   = (__bf16*)(ws + 65536 + 524288 + 262144 + 4194304); // 64 KB

    k_metric  <<<  64, 256, 0, stream>>>(mt, metric);
    k_xm      <<< 512, 256, 0, stream>>>(x, metric, xmf, xmb);
    k_swz_conn<<<8192, 256, 0, stream>>>(conn, connsw);
    k_swz_pw  <<< 128, 256, 0, stream>>>(pw, pwsw);
    k_main    <<<  64, 256, 0, stream>>>(xmf, xmb, connsw, pwsw, pb, out);
}
